// MultiHeadAttention_46978352283877
// MI455X (gfx1250) — matrix operand order repaired and checked by reading, not compiled, not run on hardware
//
#include <hip/hip_runtime.h>

// Problem constants (B=2, T=2048, C=2048, H=16, Dh=128)
#define BT   4096   // B*T
#define TSEQ 2048
#define CDIM 2048
#define C3   6144
#define NH   16
#define DH   128
#define LDC  132    // padded ld for the 128x128 C-dump tile (bank-conflict relief)

typedef __attribute__((ext_vector_type(16))) __bf16 v16bf;
typedef __attribute__((ext_vector_type(8)))  float  v8f;

union FragU { v16bf v; unsigned int u[8]; unsigned short s[16]; };

__device__ inline unsigned short f32_to_bf16(float f) {
  unsigned int u = __float_as_uint(f);
  u += 0x7FFFu + ((u >> 16) & 1u);   // round-to-nearest-even
  return (unsigned short)(u >> 16);
}
__device__ inline float bf16_to_f32(unsigned short s) {
  return __uint_as_float(((unsigned int)s) << 16);
}
__device__ inline v8f vzero() {
  v8f z;
#pragma unroll
  for (int i = 0; i < 8; ++i) z[i] = 0.0f;
  return z;
}
__device__ inline v8f wmma_bf16(v16bf a, v16bf b, v8f c) {
  return __builtin_amdgcn_wmma_f32_16x16x32_bf16(false, a, false, b, (short)0, c,
                                                 false, false);
}

// A fragment: 16x32 (MxK) bf16 from a row-major [m][k] tile with leading dim ld
// (elements, even). ISA layout: lanes 0-15 M=lane, K pairs per VGPR:
// v0..3 -> K={0..7}+8*hi, v4..7 -> K={16..23}+8*hi  (hi = lane>=16).
__device__ inline v16bf frag_A(const unsigned short* s, int ld) {
  int lane = threadIdx.x & 31;
  int m  = lane & 15;
  int kb = (lane >> 4) << 3;
  FragU f;
#pragma unroll
  for (int v = 0; v < 8; ++v) {
    int k = kb + ((v & 3) << 1) + ((v >> 2) << 4);
    f.u[v] = *(const unsigned int*)(s + m * ld + k);
  }
  return f.v;
}

// B fragment: 32x16 (KxN) bf16 from a TRANSPOSED [n][k] tile with leading dim ld.
// ISA layout: lane = N (mod 16), elems 0-7 K={0..7}+8*hi, elems 8-15 K={16..23}+8*hi
// (matches frag_A; two b128 LDS reads).
__device__ inline v16bf frag_B(const unsigned short* s, int ld) {
  int lane = threadIdx.x & 31;
  int n  = lane & 15;
  int kb = (lane >> 4) << 3;
  const unsigned short* p = s + n * ld + kb;
  FragU f;
  *(uint4*)&f.u[0] = *(const uint4*)(p);
  *(uint4*)&f.u[4] = *(const uint4*)(p + 16);
  return f.v;
}

// ---------------------------------------------------------------------------
// fp32 -> bf16 pair-packed conversion (x)
// ---------------------------------------------------------------------------
__global__ __launch_bounds__(256) void cvt_f32_bf16(const float* __restrict__ src,
                                                    unsigned int* __restrict__ dst,
                                                    int n2) {
  int i = blockIdx.x * 256 + threadIdx.x;
  if (i < n2) {
    float2 f = ((const float2*)src)[i];
    dst[i] = (unsigned int)f32_to_bf16(f.x) | ((unsigned int)f32_to_bf16(f.y) << 16);
  }
}

// ---------------------------------------------------------------------------
// fp32 (K x N, row-major) -> bf16 TRANSPOSED (N x K, row-major).
// 64x64 LDS tile, padded ld=66 so the column reads spread across banks.
// ---------------------------------------------------------------------------
#define TLD 66
__global__ __launch_bounds__(256) void cvt_transpose_f32_bf16(
    const float* __restrict__ src, unsigned short* __restrict__ dst, int K, int N) {
  __shared__ unsigned short tile[64 * TLD];
  int k0 = blockIdx.x * 64;
  int n0 = blockIdx.y * 64;
  int tid = threadIdx.x;
#pragma unroll
  for (int c = 0; c < 4; ++c) {
    int q  = tid + (c << 8);
    int k  = q >> 4;
    int no = (q & 15) << 2;
    float4 f = *(const float4*)(src + (size_t)(k0 + k) * N + n0 + no);
    unsigned short* p = &tile[k * TLD + no];
    p[0] = f32_to_bf16(f.x); p[1] = f32_to_bf16(f.y);
    p[2] = f32_to_bf16(f.z); p[3] = f32_to_bf16(f.w);
  }
  __syncthreads();
#pragma unroll
  for (int c = 0; c < 4; ++c) {
    int q  = tid + (c << 8);
    int n  = q >> 4;
    int ko = (q & 15) << 2;
    uint2 v;
    v.x = (unsigned int)tile[(ko + 0) * TLD + n] |
          ((unsigned int)tile[(ko + 1) * TLD + n] << 16);
    v.y = (unsigned int)tile[(ko + 2) * TLD + n] |
          ((unsigned int)tile[(ko + 3) * TLD + n] << 16);
    *(uint2*)(dst + (size_t)(n0 + n) * K + k0 + ko) = v;
  }
}

// ---------------------------------------------------------------------------
// Software-pipelined 128x128 GEMM mainloop over K=CDIM, bf16 WMMA, f32 acc.
// A: [m][K]; Bt: [n][K] (weights pre-transposed); both ld = CDIM.
// Double-buffered LDS (2 x (8KB A + 8KB B)); per K-step:
//   issue next-tile global_load_b128s -> run 8 WMMAs on current buffer
//   -> ds_store next buffer -> one barrier.
// ---------------------------------------------------------------------------
__device__ inline void gemm_mainloop(const unsigned short* __restrict__ Abase,
                                     const unsigned short* __restrict__ Btbase,
                                     unsigned short* sA, unsigned short* sBt,
                                     v8f (&acc)[2][4]) {
  int tid  = threadIdx.x;
  int wave = tid >> 5;
  int wm = (wave & 3) * 32;
  int wn = (wave >> 2) * 64;

  int q0 = tid,       r0 = q0 >> 2, ko0 = (q0 & 3) << 3;
  int q1 = tid + 256, r1 = q1 >> 2, ko1 = (q1 & 3) << 3;

  // Prologue: stage tile 0 into buffer 0
  *(uint4*)(sA  + r0 * 32 + ko0) = *(const uint4*)(Abase  + (size_t)r0 * CDIM + ko0);
  *(uint4*)(sA  + r1 * 32 + ko1) = *(const uint4*)(Abase  + (size_t)r1 * CDIM + ko1);
  *(uint4*)(sBt + r0 * 32 + ko0) = *(const uint4*)(Btbase + (size_t)r0 * CDIM + ko0);
  *(uint4*)(sBt + r1 * 32 + ko1) = *(const uint4*)(Btbase + (size_t)r1 * CDIM + ko1);
  __syncthreads();

  for (int kk = 0; kk < CDIM; kk += 32) {
    int curo = ((kk >> 5) & 1) * 4096;   // elements: 128*32 per buffer
    int nxto = curo ^ 4096;
    bool hasNext = (kk + 32 < CDIM);

    uint4 a0v{}, a1v{}, b0v{}, b1v{};
    if (hasNext) {  // issue next tile's loads before the math
      a0v = *(const uint4*)(Abase  + (size_t)r0 * CDIM + kk + 32 + ko0);
      a1v = *(const uint4*)(Abase  + (size_t)r1 * CDIM + kk + 32 + ko1);
      b0v = *(const uint4*)(Btbase + (size_t)r0 * CDIM + kk + 32 + ko0);
      b1v = *(const uint4*)(Btbase + (size_t)r1 * CDIM + kk + 32 + ko1);
    }
    if (kk + 64 < CDIM) {  // L2-distance prefetch two tiles ahead
      __builtin_prefetch(Abase  + (size_t)r0 * CDIM + kk + 64 + ko0, 0, 1);
      __builtin_prefetch(Btbase + (size_t)r0 * CDIM + kk + 64 + ko0, 0, 1);
    }

    v16bf a0 = frag_A(sA + curo + (wm + 0)  * 32, 32);
    v16bf a1 = frag_A(sA + curo + (wm + 16) * 32, 32);
#pragma unroll
    for (int j = 0; j < 4; ++j) {
      v16bf b = frag_B(sBt + curo + (wn + j * 16) * 32, 32);
      acc[0][j] = wmma_bf16(a0, b, acc[0][j]);
      acc[1][j] = wmma_bf16(a1, b, acc[1][j]);
    }

    if (hasNext) {  // loadcnt wait lands here, after the WMMA block
      *(uint4*)(sA  + nxto + r0 * 32 + ko0) = a0v;
      *(uint4*)(sA  + nxto + r1 * 32 + ko1) = a1v;
      *(uint4*)(sBt + nxto + r0 * 32 + ko0) = b0v;
      *(uint4*)(sBt + nxto + r1 * 32 + ko1) = b1v;
    }
    __syncthreads();
  }
}

// ---------------------------------------------------------------------------
// qkv = x @ W_qkv, then RoPE on Q/K (1/sqrt(Dh) folded into Q).
// Q,K -> bf16 (B,H,T,Dh); V -> bf16 TRANSPOSED (B,H,Dh,T).
// grid = (BT/128, 48); ntile 0-15 Q heads, 16-31 K heads, 32-47 V heads.
// ---------------------------------------------------------------------------
__global__ __launch_bounds__(256) void qkv_rope_kernel(
    const unsigned short* __restrict__ xb, const unsigned short* __restrict__ wqkvT,
    unsigned short* __restrict__ qws, unsigned short* __restrict__ kws,
    unsigned short* __restrict__ vws) {
  extern __shared__ unsigned char smem[];
  unsigned short* sA  = (unsigned short*)smem;             // 2 x 128x32
  unsigned short* sBt = (unsigned short*)(smem + 16384);   // 2 x 128x32 [n][k]
  unsigned short* sCb = (unsigned short*)smem;             // 128xLDC (reused)

  int mtile = blockIdx.x, ntile = blockIdx.y;
  int tid = threadIdx.x, lane = tid & 31, wave = tid >> 5;
  int wm = (wave & 3) * 32, wn = (wave >> 2) * 64;
  int hi = lane >> 4, nn = lane & 15;

  v8f acc[2][4];
#pragma unroll
  for (int a = 0; a < 2; ++a)
#pragma unroll
    for (int b = 0; b < 4; ++b) acc[a][b] = vzero();

  gemm_mainloop(xb + (size_t)mtile * 128 * CDIM,
                wqkvT + (size_t)ntile * 128 * CDIM, sA, sBt, acc);

  // Dump C tile (bf16) to LDS for cross-wave RoPE pairing / V transpose
#pragma unroll
  for (int mt = 0; mt < 2; ++mt)
#pragma unroll
    for (int j = 0; j < 4; ++j)
#pragma unroll
      for (int r = 0; r < 8; ++r) {
        int m = wm + mt * 16 + r + hi * 8;
        int n = wn + j * 16 + nn;
        sCb[m * LDC + n] = f32_to_bf16(acc[mt][j][r]);
      }
  __syncthreads();

  int h = ntile & 15;
  int which = ntile >> 4;             // 0=Q, 1=K, 2=V
  int b = mtile >> 4;                 // 16 m-tiles per batch
  int t0 = (mtile & 15) * 128;

  if (which == 2) {
    // V: transpose to (B,H,Dh,T)
    unsigned short* dstv = vws + ((size_t)b * NH + h) * DH * TSEQ + t0;
    for (int i = tid; i < 128 * 128; i += 256) {
      int d = i >> 7, m = i & 127;
      dstv[(size_t)d * TSEQ + m] = sCb[m * LDC + d];
    }
  } else {
    unsigned short* dst =
        ((which == 0) ? qws : kws) + (((size_t)b * NH + h) * TSEQ + t0) * DH;
    float qs = (which == 0) ? 0.08838834764831845f : 1.0f;  // 1/sqrt(Dh) into Q
    for (int i = tid; i < 128 * 64; i += 256) {
      int m = i >> 6;
      int j = i & 63;
      float x1 = bf16_to_f32(sCb[m * LDC + j]);
      float x2 = bf16_to_f32(sCb[m * LDC + j + 64]);
      float ang = (float)(t0 + m) * __powf(10000.0f, -(float)j * (1.0f / 64.0f));
      float sn, cs;
      __sincosf(ang, &sn, &cs);
      dst[m * 128 + j]      = f32_to_bf16((x1 * cs - x2 * sn) * qs);
      dst[m * 128 + j + 64] = f32_to_bf16((x1 * sn + x2 * cs) * qs);
    }
  }
}

// ---------------------------------------------------------------------------
// Flash attention. grid = (T/128, B*H). 256 threads = 8 waves; wave w owns
// q-rows [w*16, w*16+16). Key tiles of 64; causal loop kt <= 2*qt+1.
// Register-prefetched staging: next K/V tiles load while current tile computes.
// ---------------------------------------------------------------------------
__global__ __launch_bounds__(256) void flash_attn_kernel(
    const unsigned short* __restrict__ qws, const unsigned short* __restrict__ kws,
    const unsigned short* __restrict__ vws, unsigned short* __restrict__ aws) {
  extern __shared__ unsigned char smem[];
  unsigned short* sK = (unsigned short*)smem;            // [key<64][d<128]
  unsigned short* sV = (unsigned short*)(smem + 16384);  // [d<128][key<64]
  unsigned short* sP = (unsigned short*)(smem + 32768);  // 8 waves x 16x64

  int qt = blockIdx.x, bh = blockIdx.y;
  int tid = threadIdx.x, lane = tid & 31, wave = tid >> 5;
  int hi = lane >> 4, nn = lane & 15;

  const unsigned short* Q  = qws + (size_t)bh * TSEQ * DH;
  const unsigned short* K  = kws + (size_t)bh * TSEQ * DH;
  const unsigned short* Vt = vws + (size_t)bh * DH * TSEQ;  // [d][t]

  // Prologue: issue kt=0 K/V loads, then Q fragment gathers (all in flight)
  uint4 kreg[4], vreg[4];
#pragma unroll
  for (int c = 0; c < 4; ++c) {
    int q = tid + (c << 8);
    kreg[c] = *(const uint4*)(K + (size_t)(q >> 4) * DH + ((q & 15) << 3));
    vreg[c] = *(const uint4*)(Vt + (size_t)(q >> 3) * TSEQ + ((q & 7) << 3));
  }
  v16bf qa[4];
  {
    const unsigned short* Qb = Q + (size_t)(qt * 128 + wave * 16) * DH;
#pragma unroll
    for (int kk = 0; kk < 4; ++kk) qa[kk] = frag_A(Qb + kk * 32, DH);
  }

  v8f o[8];
#pragma unroll
  for (int j = 0; j < 8; ++j) o[j] = vzero();
  float rmax[8], rsum[8];
#pragma unroll
  for (int r = 0; r < 8; ++r) { rmax[r] = -3.0e38f; rsum[r] = 0.0f; }

  unsigned short* pw = sP + wave * 16 * 64;
  int ktEnd = 2 * qt + 1;
  for (int kt = 0; kt <= ktEnd; ++kt) {
    // Dump the prefetched tile into LDS
#pragma unroll
    for (int c = 0; c < 4; ++c) {
      int q = tid + (c << 8);
      *(uint4*)(sK + (q >> 4) * 128 + ((q & 15) << 3)) = kreg[c];
      *(uint4*)(sV + (q >> 3) * 64  + ((q & 7)  << 3)) = vreg[c];
    }
    __syncthreads();

    // Issue next tile's loads; latency covered by S/softmax/PV below
    if (kt < ktEnd) {
#pragma unroll
      for (int c = 0; c < 4; ++c) {
        int q = tid + (c << 8);
        kreg[c] = *(const uint4*)(K + (size_t)((kt + 1) * 64 + (q >> 4)) * DH +
                                  ((q & 15) << 3));
        vreg[c] = *(const uint4*)(Vt + (size_t)(q >> 3) * TSEQ + (kt + 1) * 64 +
                                  ((q & 7) << 3));
      }
    }

    // S = Q K^T  (4 key sub-tiles x 4 Dh k-steps)
    v8f s[4];
#pragma unroll
    for (int j = 0; j < 4; ++j) {
      s[j] = vzero();
#pragma unroll
      for (int kk = 0; kk < 4; ++kk) {
        v16bf bk = frag_B(sK + (j * 16) * 128 + kk * 32, 128);
        s[j] = wmma_bf16(qa[kk], bk, s[j]);
      }
    }

    // Causal mask on the (at most two) diagonal tiles
    if (kt >= 2 * qt) {
      int qbase = qt * 128 + wave * 16 + hi * 8;
#pragma unroll
      for (int j = 0; j < 4; ++j) {
        int key = kt * 64 + j * 16 + nn;
#pragma unroll
        for (int r = 0; r < 8; ++r)
          if (key > qbase + r) s[j][r] = -3.0e38f;
      }
    }

    // Online softmax: row stats replicated across each half-wave
#pragma unroll
    for (int r = 0; r < 8; ++r) {
      float m = s[0][r];
#pragma unroll
      for (int j = 1; j < 4; ++j) m = fmaxf(m, s[j][r]);
#pragma unroll
      for (int d = 1; d < 16; d <<= 1) m = fmaxf(m, __shfl_xor(m, d, 32));
      float nm = fmaxf(rmax[r], m);
      float alpha = __expf(rmax[r] - nm);
      rmax[r] = nm;
      float ps = 0.0f;
#pragma unroll
      for (int j = 0; j < 4; ++j) {
        float p = __expf(s[j][r] - nm);
        s[j][r] = p;
        ps += p;
      }
#pragma unroll
      for (int d = 1; d < 16; d <<= 1) ps += __shfl_xor(ps, d, 32);
      rsum[r] = rsum[r] * alpha + ps;
#pragma unroll
      for (int j2 = 0; j2 < 8; ++j2) o[j2][r] *= alpha;
    }

    // P -> wave-private LDS (C-layout -> A-layout shuffle via LDS)
#pragma unroll
    for (int j = 0; j < 4; ++j)
#pragma unroll
      for (int r = 0; r < 8; ++r)
        pw[(r + hi * 8) * 64 + j * 16 + nn] = f32_to_bf16(s[j][r]);
    asm volatile("s_wait_dscnt 0" ::: "memory");  // wave's P stores before reloads

    // O += P @ V
#pragma unroll
    for (int kk = 0; kk < 2; ++kk) {
      v16bf pa = frag_A(pw + kk * 32, 64);
#pragma unroll
      for (int j2 = 0; j2 < 8; ++j2) {
        v16bf vb = frag_B(sV + (j2 * 16) * 64 + kk * 32, 64);
        o[j2] = wmma_bf16(pa, vb, o[j2]);
      }
    }
    __syncthreads();
  }

  // Normalize and write attention output, (B*T, C) bf16, head-major columns
  int b = bh >> 4, h = bh & 15;
  unsigned short* dst =
      aws + (size_t)(b * TSEQ + qt * 128 + wave * 16) * CDIM + h * DH;
#pragma unroll
  for (int r = 0; r < 8; ++r) {
    float inv = 1.0f / rsum[r];
    int m = r + hi * 8;
#pragma unroll
    for (int j2 = 0; j2 < 8; ++j2)
      dst[(size_t)m * CDIM + j2 * 16 + nn] = f32_to_bf16(o[j2][r] * inv);
  }
}

// ---------------------------------------------------------------------------
// out = attn @ W_out (W_out pre-transposed), f32 out. grid = (BT/128, 16).
// ---------------------------------------------------------------------------
__global__ __launch_bounds__(256) void out_proj_kernel(
    const unsigned short* __restrict__ aws, const unsigned short* __restrict__ woutT,
    float* __restrict__ out) {
  extern __shared__ unsigned char smem[];
  unsigned short* sA  = (unsigned short*)smem;
  unsigned short* sBt = (unsigned short*)(smem + 16384);

  int mtile = blockIdx.x, ntile = blockIdx.y;
  int tid = threadIdx.x, lane = tid & 31, wave = tid >> 5;
  int wm = (wave & 3) * 32, wn = (wave >> 2) * 64;
  int hi = lane >> 4, nn = lane & 15;

  v8f acc[2][4];
#pragma unroll
  for (int a = 0; a < 2; ++a)
#pragma unroll
    for (int b = 0; b < 4; ++b) acc[a][b] = vzero();

  gemm_mainloop(aws + (size_t)mtile * 128 * CDIM,
                woutT + (size_t)ntile * 128 * CDIM, sA, sBt, acc);

#pragma unroll
  for (int mt = 0; mt < 2; ++mt)
#pragma unroll
    for (int j = 0; j < 4; ++j)
#pragma unroll
      for (int r = 0; r < 8; ++r) {
        int m = mtile * 128 + wm + mt * 16 + r + hi * 8;
        int n = ntile * 128 + wn + j * 16 + nn;
        out[(size_t)m * CDIM + n] = acc[mt][j][r];
      }
}

// ---------------------------------------------------------------------------
// Host launcher
// ---------------------------------------------------------------------------
extern "C" void kernel_launch(void* const* d_in, const int* in_sizes, int n_in,
                              void* d_out, int out_size, void* d_ws, size_t ws_size,
                              hipStream_t stream) {
  const float* x    = (const float*)d_in[0];
  const float* Wqkv = (const float*)d_in[1];
  const float* Wout = (const float*)d_in[2];
  float* out = (float*)d_out;

  // Workspace layout (bf16 as unsigned short), ~117.5 MB total
  unsigned short* xb     = (unsigned short*)d_ws;
  unsigned short* wqkvT  = xb    + (size_t)BT * CDIM;    // (3C x C) transposed
  unsigned short* woutT  = wqkvT + (size_t)CDIM * C3;    // (C x C) transposed
  unsigned short* qws    = woutT + (size_t)CDIM * CDIM;  // (B,H,T,Dh)
  unsigned short* kws    = qws   + (size_t)BT * CDIM;    // (B,H,T,Dh)
  unsigned short* vws    = kws   + (size_t)BT * CDIM;    // (B,H,Dh,T)
  unsigned short* aws    = vws   + (size_t)BT * CDIM;    // attn out (B*T, C)

  int n2x = (BT * CDIM) / 2;
  cvt_f32_bf16<<<dim3((n2x + 255) / 256), dim3(256), 0, stream>>>(
      x, (unsigned int*)xb, n2x);
  cvt_transpose_f32_bf16<<<dim3(CDIM / 64, C3 / 64), dim3(256), 0, stream>>>(
      Wqkv, wqkvT, CDIM, C3);
  cvt_transpose_f32_bf16<<<dim3(CDIM / 64, CDIM / 64), dim3(256), 0, stream>>>(
      Wout, woutT, CDIM, CDIM);

  qkv_rope_kernel<<<dim3(BT / 128, 48), dim3(256), 128 * LDC * 2, stream>>>(
      xb, wqkvT, qws, kws, vws);
  flash_attn_kernel<<<dim3(TSEQ / 128, 2 * NH), dim3(256), 49152, stream>>>(
      qws, kws, vws, aws);
  out_proj_kernel<<<dim3(BT / 128, CDIM / 128), dim3(256), 32768, stream>>>(
      aws, woutT, out);
}
